// LinearTransformerCausalDecoder_42691974922503
// MI455X (gfx1250) — matrix operand order repaired and checked by reading, not compiled, not run on hardware
//
#include <hip/hip_runtime.h>

typedef __bf16 bf16_t;
typedef __attribute__((ext_vector_type(4)))  float  vf4;
typedef __attribute__((ext_vector_type(8)))  float  v8f;
typedef __attribute__((ext_vector_type(16))) __bf16 v16bf;
typedef __attribute__((ext_vector_type(4)))  unsigned u32x4;
typedef __attribute__((ext_vector_type(8)))  int      i32x8;
typedef __attribute__((ext_vector_type(4)))  int      i32x4;

#if defined(__gfx1250__) && __has_builtin(__builtin_amdgcn_tensor_load_to_lds)
#define USE_TDM 1
#else
#define USE_TDM 0
#endif

union AB32 { struct { vf4 lo, hi; } f; v16bf v; };

__device__ inline v8f zero8() { v8f z; for (int i = 0; i < 8; ++i) z[i] = 0.0f; return z; }

// A-fragment (16x32 bf16, MxK): lanes 0-15: row=lane, K 0..15; lanes 16-31: K 16..31
__device__ inline v16bf frag_a(const bf16_t* row, int kt, int half) {
  AB32 u;
  u.f.lo = *(const vf4*)(row + kt * 32 + half * 16);
  u.f.hi = *(const vf4*)(row + kt * 32 + half * 16 + 8);
  return u.v;
}
// B-fragment from transposed LDS tile Bt[n][k]: lanes 0-15: n=lane, K 0..15; lanes 16-31: K 16..31
__device__ inline v16bf frag_b(const bf16_t* row, int kt, int half) {
  return *(const v16bf*)(row + kt * 32 + half * 16);
}
__device__ inline v8f wmma_bf16(v16bf a, v16bf b, v8f c) {
  return __builtin_amdgcn_wmma_f32_16x16x32_bf16(false, a, false, b, (short)0, c, false, false);
}

#define ACT_NONE 0
#define ACT_ELU1 1
#define ACT_GELU 2

// ---------------------------------------------------------------------------
// f32 -> bf16 conversion (weights / inputs), grid-stride
// ---------------------------------------------------------------------------
__global__ void cvt_f32_bf16_k(const float* __restrict__ in, bf16_t* __restrict__ out, int n) {
  for (int i = blockIdx.x * blockDim.x + threadIdx.x; i < n; i += gridDim.x * blockDim.x)
    out[i] = (bf16_t)in[i];
}

// ---------------------------------------------------------------------------
// GEMM: C[2048,N] = act(A[2048,K] x W[K,N] + bias). 128x128 block tile,
// 256 threads (8 waves, 2x4), each wave 4x2 wmma tiles, K step 64
// (two wmma k-steps per LDS stage). A tile staged by the Tensor Data Mover.
// ---------------------------------------------------------------------------
__global__ __launch_bounds__(256) void gemm_bf16_k(
    const bf16_t* __restrict__ A, const bf16_t* __restrict__ W,
    const float* __restrict__ bias, bf16_t* __restrict__ C,
    int K, int N, int act)
{
  __shared__ __align__(64) bf16_t sA[128][64];    // A tile row-major [m][k]
  __shared__ __align__(64) bf16_t sBt[128][64];   // W tile transposed [n][k]

  const int tid = threadIdx.x;
  const int lane = tid & 31, wave = tid >> 5;
  const int half = lane >> 4, lr = lane & 15;
  const int waveM = wave >> 2, waveN = wave & 3;
  const int blockM = blockIdx.x * 128, blockN = blockIdx.y * 128;

  v8f acc[4][2];
  for (int mi = 0; mi < 4; ++mi)
    for (int ni = 0; ni < 2; ++ni) acc[mi][ni] = zero8();

  const int wk = tid & 63, wc = tid >> 6;         // W loader: k-row, n-group

#if USE_TDM
  // Tensor DMA descriptor (2-D): tile 64 elem x 128 rows of bf16 from A.
  i32x8 g1;
  g1[0] = 0x00010000;                         // data_size = 1 (2 bytes/elem)
  g1[1] = (int)((K & 0xFFFF) << 16);          // tensor_dim0 lo16
  g1[2] = (int)((unsigned)K >> 16) | (int)((2048u & 0xFFFF) << 16); // dim0 hi | dim1 lo
  g1[3] = (int)(64u << 16);                   // dim1 hi (0) | tile_dim0 = 64
  g1[4] = 128;                                // tile_dim1 = 128, tile_dim2 = 0
  g1[5] = K;                                  // tensor_dim0_stride lo32
  g1[6] = 0; g1[7] = 0;
  i32x4 gz; gz[0] = 0; gz[1] = 0; gz[2] = 0; gz[3] = 0;
  const unsigned ldsA = (unsigned)(size_t)(&sA[0][0]);
  const unsigned long long gA =
      (unsigned long long)(size_t)(A + (size_t)blockM * K);
#else
  const int ar = tid >> 1, akq = (tid & 1) * 32;  // A loader fallback
#endif

  for (int k0 = 0; k0 < K; k0 += 64) {
#if USE_TDM
    if (wave == 0) { // one TDM op stages the whole 128x64 A tile into LDS
      const unsigned long long ga = gA + (unsigned long long)k0 * 2u;
      u32x4 g0;
      g0[0] = 1u;                                      // count = 1 descriptor
      g0[1] = ldsA;                                    // lds_addr
      g0[2] = (unsigned)ga;                            // global_addr[31:0]
      g0[3] = (unsigned)((ga >> 32) & 0x01FFFFFFu) | (2u << 30); // addr hi | type=2
#if __clang_major__ >= 23
      i32x8 z8; for (int j = 0; j < 8; ++j) z8[j] = 0;
      __builtin_amdgcn_tensor_load_to_lds(g0, g1, gz, gz, z8, 0);
#else
      __builtin_amdgcn_tensor_load_to_lds(g0, g1, gz, gz, 0);
#endif
      __builtin_amdgcn_s_wait_tensorcnt(0);
    }
#else
    { // A tile -> LDS row-major (manual fallback)
      const vf4* src = (const vf4*)(A + (size_t)(blockM + ar) * K + k0 + akq);
      vf4* dst = (vf4*)&sA[ar][akq];
      dst[0] = src[0]; dst[1] = src[1]; dst[2] = src[2]; dst[3] = src[3];
    }
#endif
    { // W tile -> LDS transposed
      const bf16_t* src = W + (size_t)(k0 + wk) * N + blockN + wc * 32;
      __align__(16) bf16_t tmp[32];
      ((vf4*)tmp)[0] = ((const vf4*)src)[0];
      ((vf4*)tmp)[1] = ((const vf4*)src)[1];
      ((vf4*)tmp)[2] = ((const vf4*)src)[2];
      ((vf4*)tmp)[3] = ((const vf4*)src)[3];
      for (int j = 0; j < 32; ++j) sBt[wc * 32 + j][wk] = tmp[j];
      if (k0 + 64 < K)
        __builtin_prefetch(W + (size_t)(k0 + 64 + wk) * N + blockN + wc * 32, 0, 1);
    }
    __syncthreads();
    for (int kt = 0; kt < 2; ++kt) {
      v16bf bfr[2];
      for (int ni = 0; ni < 2; ++ni)
        bfr[ni] = frag_b(&sBt[waveN * 32 + ni * 16 + lr][0], kt, half);
      for (int mi = 0; mi < 4; ++mi) {
        v16bf af = frag_a(&sA[waveM * 64 + mi * 16 + lr][0], kt, half);
        for (int ni = 0; ni < 2; ++ni)
          acc[mi][ni] = wmma_bf16(af, bfr[ni], acc[mi][ni]);
      }
    }
    __syncthreads();
  }

  for (int mi = 0; mi < 4; ++mi)
    for (int ni = 0; ni < 2; ++ni) {
      const int col = blockN + waveN * 32 + ni * 16 + lr;
      const float bv = bias[col];
      for (int r = 0; r < 8; ++r) {
        const int row = blockM + waveM * 64 + mi * 16 + r + 8 * half;
        float x = acc[mi][ni][r] + bv;
        if (act == ACT_ELU1)      x = x > 0.0f ? x + 1.0f : __expf(x);
        else if (act == ACT_GELU) x = 0.5f * x * (1.0f + erff(x * 0.70710678118f));
        C[(size_t)row * N + col] = (bf16_t)x;
      }
    }
}

// ---------------------------------------------------------------------------
// Causal linear attention, chunked (T=64). One block per (b,h), 4 waves.
// State KV^T (f32) + z (f32) resident in LDS across 16 chunks.
// ---------------------------------------------------------------------------
__global__ __launch_bounds__(128) void causal_attn_k(
    const bf16_t* __restrict__ q, const bf16_t* __restrict__ k,
    const bf16_t* __restrict__ v, bf16_t* __restrict__ out)
{
  __shared__ __align__(64) bf16_t sQ[64][64];    // [pos][feat]
  __shared__ __align__(64) bf16_t sK[64][64];    // [pos][feat]
  __shared__ __align__(64) bf16_t sKT[64][64];   // [feat][pos]
  __shared__ __align__(64) bf16_t sVT[64][64];   // [vfeat][pos]
  __shared__ __align__(64) bf16_t sS[64][64];    // masked QK^T [qpos][kpos]
  __shared__ __align__(64) float  sKVT[64][64];  // state (V^T K) [vfeat][feat]
  __shared__ float sZ[64];
  __shared__ float sDen[64];

  const int tid = threadIdx.x;
  const int lane = tid & 31, wave = tid >> 5;
  const int half = lane >> 4, lr = lane & 15;
  const int b = blockIdx.x >> 3, h = blockIdx.x & 7;
  const size_t hoff = (size_t)h * 64;
  const int pr = tid >> 1, co = (tid & 1) * 32;

  for (int i = tid; i < 64 * 64; i += 128) (&sKVT[0][0])[i] = 0.0f;
  if (tid < 64) sZ[tid] = 0.0f;

  for (int c = 0; c < 16; ++c) {
    __syncthreads();
    { // load chunk (q row-major; k row-major + transposed; v transposed)
      const size_t gbase = ((size_t)b * 1024 + c * 64 + pr) * 512 + hoff + co;
      const vf4* gq = (const vf4*)(q + gbase);
      vf4* dq = (vf4*)&sQ[pr][co];
      dq[0] = gq[0]; dq[1] = gq[1]; dq[2] = gq[2]; dq[3] = gq[3];

      __align__(16) bf16_t tmp[32];
      const vf4* gk = (const vf4*)(k + gbase);
      vf4 a0 = gk[0], a1 = gk[1], a2 = gk[2], a3 = gk[3];
      vf4* dk = (vf4*)&sK[pr][co];
      dk[0] = a0; dk[1] = a1; dk[2] = a2; dk[3] = a3;
      ((vf4*)tmp)[0] = a0; ((vf4*)tmp)[1] = a1; ((vf4*)tmp)[2] = a2; ((vf4*)tmp)[3] = a3;
      for (int j = 0; j < 32; ++j) sKT[co + j][pr] = tmp[j];

      const vf4* gv = (const vf4*)(v + gbase);
      ((vf4*)tmp)[0] = gv[0]; ((vf4*)tmp)[1] = gv[1];
      ((vf4*)tmp)[2] = gv[2]; ((vf4*)tmp)[3] = gv[3];
      for (int j = 0; j < 32; ++j) sVT[co + j][pr] = tmp[j];
    }
    __syncthreads();

    { // S = Q K^T, causal-masked -> sS (wave w owns q-rows 16w..16w+15)
      v8f sacc[4];
      for (int nt = 0; nt < 4; ++nt) sacc[nt] = zero8();
      for (int kt = 0; kt < 2; ++kt) {
        v16bf af = frag_a(&sQ[wave * 16 + lr][0], kt, half);
        for (int nt = 0; nt < 4; ++nt)
          sacc[nt] = wmma_bf16(af, frag_b(&sK[nt * 16 + lr][0], kt, half), sacc[nt]);
      }
      for (int nt = 0; nt < 4; ++nt)
        for (int r = 0; r < 8; ++r) {
          const int qi = wave * 16 + r + 8 * half;
          const int kj = nt * 16 + lr;
          sS[qi][kj] = (bf16_t)(kj <= qi ? sacc[nt][r] : 0.0f);
        }
    }
    __syncthreads();

    if (tid < 64) { // den = rowsum(S_masked) + q . z_prev + eps
      float d = 1e-6f;
      for (int j = 0; j < 64; ++j) d += (float)sS[tid][j];
      for (int f = 0; f < 64; ++f) d += (float)sQ[tid][f] * sZ[f];
      sDen[tid] = d;
    }
    __syncthreads();

    { // num = Q * KV_prev + S_masked * V
      v8f nacc[4];
      for (int nt = 0; nt < 4; ++nt) nacc[nt] = zero8();
      for (int kt = 0; kt < 2; ++kt) {
        v16bf af = frag_a(&sQ[wave * 16 + lr][0], kt, half);
        for (int nt = 0; nt < 4; ++nt) {
          v16bf bb;  // B-fragment from f32 state, converted on the fly
          const float* src = &sKVT[nt * 16 + lr][kt * 32 + half * 16];
          for (int j = 0; j < 16; ++j) bb[j] = (bf16_t)src[j];
          nacc[nt] = wmma_bf16(af, bb, nacc[nt]);
        }
      }
      for (int kt = 0; kt < 2; ++kt) {
        v16bf af = frag_a(&sS[wave * 16 + lr][0], kt, half);
        for (int nt = 0; nt < 4; ++nt)
          nacc[nt] = wmma_bf16(af, frag_b(&sVT[nt * 16 + lr][0], kt, half), nacc[nt]);
      }
      for (int nt = 0; nt < 4; ++nt)
        for (int r = 0; r < 8; ++r) {
          const int qi = wave * 16 + r + 8 * half;
          const int vfi = nt * 16 + lr;
          out[((size_t)b * 1024 + c * 64 + qi) * 512 + hoff + vfi] =
              (bf16_t)(nacc[nt][r] / sDen[qi]);
        }
    }
    __syncthreads();

    { // state update: KV^T += V^T K (accumulate into f32 LDS); z += colsum(K)
      for (int nt = 0; nt < 4; ++nt) {
        v8f u;
        for (int r = 0; r < 8; ++r) u[r] = sKVT[wave * 16 + r + 8 * half][nt * 16 + lr];
        for (int kt = 0; kt < 2; ++kt)
          u = wmma_bf16(frag_a(&sVT[wave * 16 + lr][0], kt, half),
                        frag_b(&sKT[nt * 16 + lr][0], kt, half), u);
        for (int r = 0; r < 8; ++r) sKVT[wave * 16 + r + 8 * half][nt * 16 + lr] = u[r];
      }
      if (tid < 64) {
        float zs = sZ[tid];
        for (int p = 0; p < 64; ++p) zs += (float)sKT[tid][p];
        sZ[tid] = zs;
      }
    }
  }
}

// ---------------------------------------------------------------------------
// Cross-attn phase 1: KV^T[b,h] = V^T K over M (accumulators live in VGPRs),
// z = colsum(K). One block per (b,h).
// ---------------------------------------------------------------------------
__global__ __launch_bounds__(128) void ca_state_k(
    const bf16_t* __restrict__ k, const bf16_t* __restrict__ v,
    float* __restrict__ kvt, float* __restrict__ z)
{
  __shared__ __align__(64) bf16_t sKT[64][64];
  __shared__ __align__(64) bf16_t sVT[64][64];
  __shared__ float sZ[64];

  const int tid = threadIdx.x;
  const int lane = tid & 31, wave = tid >> 5;
  const int half = lane >> 4, lr = lane & 15;
  const int b = blockIdx.x >> 3, h = blockIdx.x & 7;
  const size_t hoff = (size_t)h * 64;
  const int pr = tid >> 1, co = (tid & 1) * 32;

  if (tid < 64) sZ[tid] = 0.0f;
  v8f u[4];
  for (int nt = 0; nt < 4; ++nt) u[nt] = zero8();

  for (int c = 0; c < 16; ++c) {
    __syncthreads();
    {
      const size_t gbase = ((size_t)b * 1024 + c * 64 + pr) * 512 + hoff + co;
      __align__(16) bf16_t tmp[32];
      const vf4* gk = (const vf4*)(k + gbase);
      ((vf4*)tmp)[0] = gk[0]; ((vf4*)tmp)[1] = gk[1];
      ((vf4*)tmp)[2] = gk[2]; ((vf4*)tmp)[3] = gk[3];
      for (int j = 0; j < 32; ++j) sKT[co + j][pr] = tmp[j];
      const vf4* gv = (const vf4*)(v + gbase);
      ((vf4*)tmp)[0] = gv[0]; ((vf4*)tmp)[1] = gv[1];
      ((vf4*)tmp)[2] = gv[2]; ((vf4*)tmp)[3] = gv[3];
      for (int j = 0; j < 32; ++j) sVT[co + j][pr] = tmp[j];
    }
    __syncthreads();
    for (int kt = 0; kt < 2; ++kt) {
      v16bf af = frag_a(&sVT[wave * 16 + lr][0], kt, half);
      for (int nt = 0; nt < 4; ++nt)
        u[nt] = wmma_bf16(af, frag_b(&sKT[nt * 16 + lr][0], kt, half), u[nt]);
    }
    if (tid < 64) {
      float zs = sZ[tid];
      for (int p = 0; p < 64; ++p) zs += (float)sKT[tid][p];
      sZ[tid] = zs;
    }
  }
  __syncthreads();
  const size_t obase = (size_t)blockIdx.x * 4096;
  for (int nt = 0; nt < 4; ++nt)
    for (int r = 0; r < 8; ++r)
      kvt[obase + (size_t)(wave * 16 + r + 8 * half) * 64 + nt * 16 + lr] = u[nt][r];
  if (tid < 64) z[blockIdx.x * 64 + tid] = sZ[tid];
}

// ---------------------------------------------------------------------------
// Cross-attn phase 2: out = (Q KV) / (Q.z + eps). Block = (s-tile, (b,h)).
// ---------------------------------------------------------------------------
__global__ __launch_bounds__(128) void ca_apply_k(
    const bf16_t* __restrict__ q, const float* __restrict__ kvt,
    const float* __restrict__ z, bf16_t* __restrict__ out)
{
  __shared__ __align__(64) bf16_t sQ[64][64];
  __shared__ __align__(64) bf16_t sKVTb[64][64];  // [vfeat][feat] bf16
  __shared__ float sZ[64];
  __shared__ float sDen[64];

  const int tid = threadIdx.x;
  const int lane = tid & 31, wave = tid >> 5;
  const int half = lane >> 4, lr = lane & 15;
  const int st = blockIdx.x;
  const int bh = blockIdx.y;
  const int b = bh >> 3, h = bh & 7;
  const size_t hoff = (size_t)h * 64;
  const int pr = tid >> 1, co = (tid & 1) * 32;

  {
    const float* src = kvt + (size_t)bh * 4096 + (size_t)tid * 32;
    bf16_t* dst = &sKVTb[0][0] + (size_t)tid * 32;
    for (int j = 0; j < 32; ++j) dst[j] = (bf16_t)src[j];
  }
  if (tid < 64) sZ[tid] = z[bh * 64 + tid];
  {
    const size_t gbase = ((size_t)b * 1024 + st * 64 + pr) * 512 + hoff + co;
    const vf4* gq = (const vf4*)(q + gbase);
    vf4* dq = (vf4*)&sQ[pr][co];
    dq[0] = gq[0]; dq[1] = gq[1]; dq[2] = gq[2]; dq[3] = gq[3];
  }
  __syncthreads();
  if (tid < 64) {
    float d = 1e-6f;
    for (int f = 0; f < 64; ++f) d += (float)sQ[tid][f] * sZ[f];
    sDen[tid] = d;
  }
  __syncthreads();
  v8f nacc[4];
  for (int nt = 0; nt < 4; ++nt) nacc[nt] = zero8();
  for (int kt = 0; kt < 2; ++kt) {
    v16bf af = frag_a(&sQ[wave * 16 + lr][0], kt, half);
    for (int nt = 0; nt < 4; ++nt)
      nacc[nt] = wmma_bf16(af, frag_b(&sKVTb[nt * 16 + lr][0], kt, half), nacc[nt]);
  }
  for (int nt = 0; nt < 4; ++nt)
    for (int r = 0; r < 8; ++r) {
      const int qi = wave * 16 + r + 8 * half;
      const int vfi = nt * 16 + lr;
      out[((size_t)b * 1024 + st * 64 + qi) * 512 + hoff + vfi] =
          (bf16_t)(nacc[nt][r] / sDen[qi]);
    }
}

// ---------------------------------------------------------------------------
// LayerNorm(x + a) (bf16 out) and final LayerNorm (f32 out). Wave per row.
// ---------------------------------------------------------------------------
__global__ __launch_bounds__(256) void ln_res_k(
    const bf16_t* __restrict__ x, const bf16_t* __restrict__ a,
    const float* __restrict__ g, const float* __restrict__ bta,
    bf16_t* __restrict__ y)
{
  const int row = blockIdx.x * 8 + (threadIdx.x >> 5);
  const int lane = threadIdx.x & 31;
  const size_t base = (size_t)row * 512;
  float vbuf[16]; float s = 0.0f;
  for (int j = 0; j < 16; ++j) {
    const int c = lane + j * 32;
    const float t = (float)x[base + c] + (float)a[base + c];
    vbuf[j] = t; s += t;
  }
  for (int m = 16; m; m >>= 1) s += __shfl_xor(s, m, 32);
  const float mean = s * (1.0f / 512.0f);
  float vs = 0.0f;
  for (int j = 0; j < 16; ++j) { const float d = vbuf[j] - mean; vs += d * d; }
  for (int m = 16; m; m >>= 1) vs += __shfl_xor(vs, m, 32);
  const float inv = rsqrtf(vs * (1.0f / 512.0f) + 1e-5f);
  for (int j = 0; j < 16; ++j) {
    const int c = lane + j * 32;
    y[base + c] = (bf16_t)((vbuf[j] - mean) * inv * g[c] + bta[c]);
  }
}

__global__ __launch_bounds__(256) void ln_final_k(
    const bf16_t* __restrict__ x, const float* __restrict__ g,
    const float* __restrict__ bta, float* __restrict__ y)
{
  const int row = blockIdx.x * 8 + (threadIdx.x >> 5);
  const int lane = threadIdx.x & 31;
  const size_t base = (size_t)row * 512;
  float vbuf[16]; float s = 0.0f;
  for (int j = 0; j < 16; ++j) {
    const int c = lane + j * 32;
    const float t = (float)x[base + c];
    vbuf[j] = t; s += t;
  }
  for (int m = 16; m; m >>= 1) s += __shfl_xor(s, m, 32);
  const float mean = s * (1.0f / 512.0f);
  float vs = 0.0f;
  for (int j = 0; j < 16; ++j) { const float d = vbuf[j] - mean; vs += d * d; }
  for (int m = 16; m; m >>= 1) vs += __shfl_xor(vs, m, 32);
  const float inv = rsqrtf(vs * (1.0f / 512.0f) + 1e-5f);
  for (int j = 0; j < 16; ++j) {
    const int c = lane + j * 32;
    y[base + c] = (vbuf[j] - mean) * inv * g[c] + bta[c];
  }
}

// ---------------------------------------------------------------------------
extern "C" void kernel_launch(void* const* d_in, const int* in_sizes, int n_in,
                              void* d_out, int out_size, void* d_ws, size_t ws_size,
                              hipStream_t stream)
{
  (void)in_sizes; (void)n_in; (void)out_size; (void)ws_size;
  const int D = 512, FF = 2048, L = 4, NT = 2048;  // NT = B*S = B*M

  const float* memory = (const float*)d_in[0];
  const float* target = (const float*)d_in[1];
  const float* saW[4] = {(const float*)d_in[2], (const float*)d_in[4],
                         (const float*)d_in[6], (const float*)d_in[8]};
  const float* sab[4] = {(const float*)d_in[3], (const float*)d_in[5],
                         (const float*)d_in[7], (const float*)d_in[9]};
  const float* caW[4] = {(const float*)d_in[10], (const float*)d_in[12],
                         (const float*)d_in[14], (const float*)d_in[16]};
  const float* cab[4] = {(const float*)d_in[11], (const float*)d_in[13],
                         (const float*)d_in[15], (const float*)d_in[17]};
  const float* ffW1 = (const float*)d_in[18]; const float* ffb1 = (const float*)d_in[19];
  const float* ffW2 = (const float*)d_in[20]; const float* ffb2 = (const float*)d_in[21];
  const float* lng[3] = {(const float*)d_in[22], (const float*)d_in[24], (const float*)d_in[26]};
  const float* lnb[3] = {(const float*)d_in[23], (const float*)d_in[25], (const float*)d_in[27]};
  const float* lnf_g = (const float*)d_in[28]; const float* lnf_b = (const float*)d_in[29];

  char* ws = (char*)d_ws; size_t off = 0;
  auto alloc = [&](size_t bytes) -> void* {
    void* p = ws + off; off = (off + bytes + 255) & ~(size_t)255; return p;
  };

  bf16_t* wX = (bf16_t*)alloc((size_t)NT * D * 2);
  bf16_t* wQ = (bf16_t*)alloc((size_t)NT * D * 2);
  bf16_t* wK = (bf16_t*)alloc((size_t)NT * D * 2);
  bf16_t* wV = (bf16_t*)alloc((size_t)NT * D * 2);
  bf16_t* wA = (bf16_t*)alloc((size_t)NT * D * 2);
  bf16_t* wO = (bf16_t*)alloc((size_t)NT * D * 2);
  bf16_t* wF = (bf16_t*)alloc((size_t)NT * FF * 2);
  bf16_t* wMem = (bf16_t*)alloc((size_t)NT * D * 2);
  bf16_t* wSa[4]; bf16_t* wCa[4];
  for (int j = 0; j < 4; ++j) wSa[j] = (bf16_t*)alloc((size_t)L * D * D * 2);
  for (int j = 0; j < 4; ++j) wCa[j] = (bf16_t*)alloc((size_t)L * D * D * 2);
  bf16_t* wF1 = (bf16_t*)alloc((size_t)L * D * FF * 2);
  bf16_t* wF2 = (bf16_t*)alloc((size_t)L * FF * D * 2);
  float* wKVT = (float*)alloc((size_t)16 * 64 * 64 * 4);
  float* wZ   = (float*)alloc((size_t)16 * 64 * 4);

  // convert inputs + weights to bf16 once
  cvt_f32_bf16_k<<<512, 256, 0, stream>>>(target, wX, NT * D);
  cvt_f32_bf16_k<<<512, 256, 0, stream>>>(memory, wMem, NT * D);
  for (int j = 0; j < 4; ++j)
    cvt_f32_bf16_k<<<512, 256, 0, stream>>>(saW[j], wSa[j], L * D * D);
  for (int j = 0; j < 4; ++j)
    cvt_f32_bf16_k<<<512, 256, 0, stream>>>(caW[j], wCa[j], L * D * D);
  cvt_f32_bf16_k<<<512, 256, 0, stream>>>(ffW1, wF1, L * D * FF);
  cvt_f32_bf16_k<<<512, 256, 0, stream>>>(ffW2, wF2, L * FF * D);

  auto gemm = [&](const bf16_t* Ain, const bf16_t* Win, const float* bin,
                  bf16_t* Cout, int K, int N, int act) {
    dim3 g(NT / 128, N / 128);
    gemm_bf16_k<<<g, 256, 0, stream>>>(Ain, Win, bin, Cout, K, N, act);
  };

  for (int i = 0; i < L; ++i) {
    const size_t wo = (size_t)i * D * D;
    // --- causal linear self-attention ---
    gemm(wX, wSa[0] + wo, sab[0] + i * D, wQ, D, D, ACT_ELU1);
    gemm(wX, wSa[1] + wo, sab[1] + i * D, wK, D, D, ACT_ELU1);
    gemm(wX, wSa[2] + wo, sab[2] + i * D, wV, D, D, ACT_NONE);
    causal_attn_k<<<16, 128, 0, stream>>>(wQ, wK, wV, wA);
    gemm(wA, wSa[3] + wo, sab[3] + i * D, wO, D, D, ACT_NONE);
    ln_res_k<<<NT / 8, 256, 0, stream>>>(wX, wO, lng[0] + i * D, lnb[0] + i * D, wX);
    // --- linear cross-attention over memory ---
    gemm(wX,   wCa[0] + wo, cab[0] + i * D, wQ, D, D, ACT_ELU1);
    gemm(wMem, wCa[1] + wo, cab[1] + i * D, wK, D, D, ACT_ELU1);
    gemm(wMem, wCa[2] + wo, cab[2] + i * D, wV, D, D, ACT_NONE);
    ca_state_k<<<16, 128, 0, stream>>>(wK, wV, wKVT, wZ);
    ca_apply_k<<<dim3(16, 16), 128, 0, stream>>>(wQ, wKVT, wZ, wA);
    gemm(wA, wCa[3] + wo, cab[3] + i * D, wO, D, D, ACT_NONE);
    ln_res_k<<<NT / 8, 256, 0, stream>>>(wX, wO, lng[1] + i * D, lnb[1] + i * D, wX);
    // --- GELU FFN ---
    gemm(wX, wF1 + (size_t)i * D * FF, ffb1 + i * FF, wF, D, FF, ACT_GELU);
    gemm(wF, wF2 + (size_t)i * FF * D, ffb2 + i * D, wO, FF, D, ACT_NONE);
    ln_res_k<<<NT / 8, 256, 0, stream>>>(wX, wO, lng[2] + i * D, lnb[2] + i * D, wX);
  }
  ln_final_k<<<NT / 8, 256, 0, stream>>>(wX, lnf_g, lnf_b, (float*)d_out);
}
